// Quant3Linear_65403761983835
// MI455X (gfx1250) — compile-verified, matrix operand order checked by reading, not run on hardware
//
#include <hip/hip_runtime.h>

typedef __attribute__((ext_vector_type(16))) _Float16 v16h;
typedef __attribute__((ext_vector_type(8)))  _Float16 v8h;
typedef __attribute__((ext_vector_type(4)))  _Float16 v4h;
typedef __attribute__((ext_vector_type(2)))  _Float16 v2h;
typedef __attribute__((ext_vector_type(8)))  float    v8f;

#define INFEAT  4096
#define OUTFEAT 11008
#define TOKENS  2048

#define BM  128
#define BN  256
#define BK  32
#define LDK 40                    // padded LDS row stride in halves (80B): 16B-aligned rows
#define NKT (INFEAT / BK)         // 128 K-steps

// ---------------------------------------------------------------------------
// Prepass 1: x f32 -> f16  (TOKENS*INFEAT elements, 4 per thread)
// ---------------------------------------------------------------------------
__global__ __launch_bounds__(256)
void convert_x_f16(const float* __restrict__ x, _Float16* __restrict__ xh)
{
    const int i = (blockIdx.x * 256 + threadIdx.x) * 4;
    float4 v = *(const float4*)(x + i);
    v4h h;
    h.x = (_Float16)v.x; h.y = (_Float16)v.y;
    h.z = (_Float16)v.z; h.w = (_Float16)v.w;
    *(v4h*)(xh + i) = h;
}

// ---------------------------------------------------------------------------
// Prepass 2: dequant + transpose qweight -> wT[n][k] f16 = (w_int - zero[n])
// LDS-tiled 64(n) x 64(k); zero-point folded via the 0x64XX exponent trick.
// ---------------------------------------------------------------------------
#define TSTR 72   // LDS tile stride in halves (144B, 16B aligned)
__global__ __launch_bounds__(256)
void dequant_w_f16(const unsigned int* __restrict__ qweight,
                   const int* __restrict__ zeros,
                   _Float16* __restrict__ wT)
{
    __shared__ _Float16 t[64 * TSTR];

    const int tid = threadIdx.x;
    const int n0  = blockIdx.x * 64;
    const int k0  = blockIdx.y * 64;

    const int col = tid & 63;          // n within tile
    const int grp = tid >> 6;          // 4 groups of 4 packed rows
    const int gn  = n0 + col;

    const float zf = (float)zeros[gn];
    v2h zp; zp.x = (_Float16)(1024.0f + zf); zp.y = zp.x;

    const unsigned int* qp = qweight + (k0 / 4) * OUTFEAT + gn;
#pragma unroll
    for (int j = 0; j < 4; ++j) {
        const int pr = grp * 4 + j;                  // packed row in tile (0..15)
        const unsigned int q = qp[pr * OUTFEAT];
        unsigned int lo = __builtin_amdgcn_perm(q, 0x64646464u, 0x00050004u);
        unsigned int hi = __builtin_amdgcn_perm(q, 0x64646464u, 0x00070006u);
        v2h h01 = __builtin_bit_cast(v2h, lo) - zp;
        v2h h23 = __builtin_bit_cast(v2h, hi) - zp;
        uint2 st;
        st.x = __builtin_bit_cast(unsigned int, h01);
        st.y = __builtin_bit_cast(unsigned int, h23);
        *(uint2*)&t[col * TSTR + pr * 4] = st;
    }
    __syncthreads();

    // write 64 rows(n) x 64 halves(k), coalesced b128
#pragma unroll
    for (int i = 0; i < 2; ++i) {
        const int c   = tid + i * 256;               // 0..511
        const int row = c >> 3;
        const int seg = c & 7;                       // 8 x 16B chunks per row
        v8h v = *(const v8h*)&t[row * TSTR + seg * 8];
        *(v8h*)&wT[(size_t)(n0 + row) * INFEAT + k0 + seg * 8] = v;
    }
}

// ---------------------------------------------------------------------------
// Main GEMM (fast path): f16 tiles stream global -> LDS via async copies
// (no VGPR staging, no conversion in the hot loop), 16 WMMA / wave / K-step.
// ---------------------------------------------------------------------------
__global__ __launch_bounds__(256, 1)
void q8_gemm_wmma_async(const _Float16* __restrict__ xh,
                        const _Float16* __restrict__ wT,
                        const float* __restrict__ scales,
                        const float* __restrict__ bias,
                        float* __restrict__ out)
{
    __shared__ _Float16 sA[2][BM * LDK];
    __shared__ _Float16 sB[2][BN * LDK];

    const int tid   = threadIdx.x;
    const int lane  = tid & 31;
    const int wave  = tid >> 5;
    const int waveM = wave >> 2;       // 0..1 -> 64-row slab
    const int waveN = wave & 3;        // 0..3 -> 64-col slab

    const int blockN = blockIdx.x * BN;
    const int blockM = blockIdx.y * BM;

    // Async-copy mapping: 16B chunks; A tile 128x(32h)=512 chunks, B tile 256x(32h)=1024.
    auto issueStage = [&](int kt, int buf) {
        const _Float16* ga = xh + (size_t)blockM * INFEAT + kt * BK;
#pragma unroll
        for (int i = 0; i < 2; ++i) {
            const int c   = tid + i * 256;
            const int row = c >> 2;
            const int seg = c & 3;                   // 4 x 16B per row
            const _Float16* g = ga + row * INFEAT + seg * 8;
            unsigned int lds = (unsigned int)(size_t)&sA[buf][row * LDK + seg * 8];
            asm volatile("global_load_async_to_lds_b128 %0, %1, off"
                         :: "v"(lds), "v"((unsigned long long)(size_t)g) : "memory");
        }
        const _Float16* gb = wT + (size_t)blockN * INFEAT + kt * BK;
#pragma unroll
        for (int i = 0; i < 4; ++i) {
            const int c   = tid + i * 256;
            const int row = c >> 2;
            const int seg = c & 3;
            const _Float16* g = gb + row * INFEAT + seg * 8;
            unsigned int lds = (unsigned int)(size_t)&sB[buf][row * LDK + seg * 8];
            asm volatile("global_load_async_to_lds_b128 %0, %1, off"
                         :: "v"(lds), "v"((unsigned long long)(size_t)g) : "memory");
        }
    };

    v8f zero8 = {};
    v8f acc[4][4];
#pragma unroll
    for (int i = 0; i < 4; ++i)
#pragma unroll
        for (int j = 0; j < 4; ++j)
            acc[i][j] = zero8;

    const int frow = lane & 15;          // M (A) / N (B) within a 16-tile
    const int klo  = (lane >> 4) << 3;   // 0 or 8

    issueStage(0, 0);
    asm volatile("s_wait_asynccnt 0x0" ::: "memory");
    __syncthreads();

    for (int kt = 0; kt < NKT; ++kt) {
        const int cur = kt & 1;
        if (kt + 1 < NKT) issueStage(kt + 1, cur ^ 1);

        v16h aFrag[4], bFrag[4];
#pragma unroll
        for (int wm = 0; wm < 4; ++wm) {
            const _Float16* p = &sA[cur][(waveM * 64 + wm * 16 + frow) * LDK + klo];
            v8h lo = *(const v8h*)(p);
            v8h hi = *(const v8h*)(p + 16);
            aFrag[wm] = __builtin_shufflevector(lo, hi,
                0,1,2,3,4,5,6,7,8,9,10,11,12,13,14,15);
        }
#pragma unroll
        for (int wn = 0; wn < 4; ++wn) {
            const _Float16* p = &sB[cur][(waveN * 64 + wn * 16 + frow) * LDK + klo];
            v8h lo = *(const v8h*)(p);
            v8h hi = *(const v8h*)(p + 16);
            bFrag[wn] = __builtin_shufflevector(lo, hi,
                0,1,2,3,4,5,6,7,8,9,10,11,12,13,14,15);
        }

#pragma unroll
        for (int wm = 0; wm < 4; ++wm)
#pragma unroll
            for (int wn = 0; wn < 4; ++wn)
                acc[wm][wn] = __builtin_amdgcn_wmma_f32_16x16x32_f16(
                    false, aFrag[wm], false, bFrag[wn],
                    (short)0, acc[wm][wn], false, false);

        asm volatile("s_wait_asynccnt 0x0" ::: "memory");
        __syncthreads();
    }

    const int rbase = blockM + waveM * 64 + ((lane >> 4) << 3);
#pragma unroll
    for (int wn = 0; wn < 4; ++wn) {
        const int c  = blockN + waveN * 64 + wn * 16 + (lane & 15);
        const float sc = scales[c];
        const float bs = bias[c];
#pragma unroll
        for (int wm = 0; wm < 4; ++wm) {
            const int r0 = rbase + wm * 16;
#pragma unroll
            for (int r = 0; r < 8; ++r)
                out[(r0 + r) * OUTFEAT + c] = acc[wm][wn][r] * sc + bs;
        }
    }
}

// ---------------------------------------------------------------------------
// Fallback (self-contained fused kernel, round-2 version) for small ws_size
// ---------------------------------------------------------------------------
__global__ __launch_bounds__(256, 1)
void q8_gemm_wmma_fused(const float* __restrict__ x,
                        const unsigned int* __restrict__ qweight,
                        const float* __restrict__ scales,
                        const int* __restrict__ zeros,
                        const float* __restrict__ bias,
                        float* __restrict__ out)
{
    __shared__ _Float16 sA[2][BM * LDK];
    __shared__ _Float16 sB[2][BN * LDK];

    const int tid   = threadIdx.x;
    const int lane  = tid & 31;
    const int wave  = tid >> 5;
    const int waveM = wave >> 2;
    const int waveN = wave & 3;

    const int blockN = blockIdx.x * BN;
    const int blockM = blockIdx.y * BM;

    const int arow = tid >> 3;
    const int acol = (tid & 7) << 2;
    const int bn   = tid;
    const int gcol = blockN + bn;

    const float zf = (float)zeros[gcol];
    v2h zpack;
    zpack.x = (_Float16)(1024.0f + zf);
    zpack.y = zpack.x;

    float4       aReg[4];
    unsigned int bReg[8];

    auto loadStage = [&](int kt) {
        const float* xp = x + (blockM + arow) * INFEAT + kt * BK + acol;
#pragma unroll
        for (int i = 0; i < 4; ++i)
            aReg[i] = *(const float4*)(xp + 32 * i * INFEAT);
        const unsigned int* qp = qweight + (kt * (BK / 4)) * OUTFEAT + gcol;
#pragma unroll
        for (int j = 0; j < 8; ++j)
            bReg[j] = qp[j * OUTFEAT];
    };

    auto storeStage = [&](int buf) {
#pragma unroll
        for (int i = 0; i < 4; ++i) {
            v4h h;
            h.x = (_Float16)aReg[i].x; h.y = (_Float16)aReg[i].y;
            h.z = (_Float16)aReg[i].z; h.w = (_Float16)aReg[i].w;
            *(v4h*)&sA[buf][(arow + 32 * i) * LDK + acol] = h;
        }
#pragma unroll
        for (int j = 0; j < 8; ++j) {
            const unsigned int q = bReg[j];
            unsigned int lo = __builtin_amdgcn_perm(q, 0x64646464u, 0x00050004u);
            unsigned int hi = __builtin_amdgcn_perm(q, 0x64646464u, 0x00070006u);
            v2h h01 = __builtin_bit_cast(v2h, lo) - zpack;
            v2h h23 = __builtin_bit_cast(v2h, hi) - zpack;
            uint2 st;
            st.x = __builtin_bit_cast(unsigned int, h01);
            st.y = __builtin_bit_cast(unsigned int, h23);
            *(uint2*)&sB[buf][bn * LDK + 4 * j] = st;
        }
    };

    v8f zero8 = {};
    v8f acc[4][4];
#pragma unroll
    for (int i = 0; i < 4; ++i)
#pragma unroll
        for (int j = 0; j < 4; ++j)
            acc[i][j] = zero8;

    const int frow = lane & 15;
    const int klo  = (lane >> 4) << 3;

    loadStage(0);
    storeStage(0);
    __syncthreads();

    for (int kt = 0; kt < NKT; ++kt) {
        const int cur = kt & 1;
        if (kt + 1 < NKT) loadStage(kt + 1);

        v16h aFrag[4], bFrag[4];
#pragma unroll
        for (int wm = 0; wm < 4; ++wm) {
            const _Float16* p = &sA[cur][(waveM * 64 + wm * 16 + frow) * LDK + klo];
            v8h lo = *(const v8h*)(p);
            v8h hi = *(const v8h*)(p + 16);
            aFrag[wm] = __builtin_shufflevector(lo, hi,
                0,1,2,3,4,5,6,7,8,9,10,11,12,13,14,15);
        }
#pragma unroll
        for (int wn = 0; wn < 4; ++wn) {
            const _Float16* p = &sB[cur][(waveN * 64 + wn * 16 + frow) * LDK + klo];
            v8h lo = *(const v8h*)(p);
            v8h hi = *(const v8h*)(p + 16);
            bFrag[wn] = __builtin_shufflevector(lo, hi,
                0,1,2,3,4,5,6,7,8,9,10,11,12,13,14,15);
        }

#pragma unroll
        for (int wm = 0; wm < 4; ++wm)
#pragma unroll
            for (int wn = 0; wn < 4; ++wn)
                acc[wm][wn] = __builtin_amdgcn_wmma_f32_16x16x32_f16(
                    false, aFrag[wm], false, bFrag[wn],
                    (short)0, acc[wm][wn], false, false);

        if (kt + 1 < NKT) storeStage((kt + 1) & 1);
        __syncthreads();
    }

    const int rbase = blockM + waveM * 64 + ((lane >> 4) << 3);
#pragma unroll
    for (int wn = 0; wn < 4; ++wn) {
        const int c  = blockN + waveN * 64 + wn * 16 + (lane & 15);
        const float sc = scales[c];
        const float bs = bias[c];
#pragma unroll
        for (int wm = 0; wm < 4; ++wm) {
            const int r0 = rbase + wm * 16;
#pragma unroll
            for (int r = 0; r < 8; ++r)
                out[(r0 + r) * OUTFEAT + c] = acc[wm][wn][r] * sc + bs;
        }
    }
}

extern "C" void kernel_launch(void* const* d_in, const int* in_sizes, int n_in,
                              void* d_out, int out_size, void* d_ws, size_t ws_size,
                              hipStream_t stream) {
    (void)in_sizes; (void)n_in; (void)out_size;
    const float*        xp = (const float*)d_in[0];
    const unsigned int* qw = (const unsigned int*)d_in[1];
    const float*        sc = (const float*)d_in[2];
    const int*          zp = (const int*)d_in[3];
    const float*        bi = (const float*)d_in[4];
    float*              op = (float*)d_out;

    const size_t xBytes = (size_t)TOKENS * INFEAT * sizeof(_Float16);   // 16.8 MB
    const size_t wBytes = (size_t)INFEAT * OUTFEAT * sizeof(_Float16);  // 90.2 MB

    dim3 grid(OUTFEAT / BN, TOKENS / BM);   // (43, 16)

    if (ws_size >= xBytes + wBytes) {
        _Float16* xh = (_Float16*)d_ws;
        _Float16* wT = (_Float16*)((char*)d_ws + xBytes);

        convert_x_f16<<<(TOKENS * INFEAT) / (256 * 4), 256, 0, stream>>>(xp, xh);
        dim3 wgrid(OUTFEAT / 64, INFEAT / 64);   // (172, 64)
        dequant_w_f16<<<wgrid, 256, 0, stream>>>(qw, zp, wT);
        q8_gemm_wmma_async<<<grid, 256, 0, stream>>>(xh, wT, sc, bi, op);
    } else {
        q8_gemm_wmma_fused<<<grid, 256, 0, stream>>>(xp, qw, sc, zp, bi, op);
    }
}
